// Creator_32134945309019
// MI455X (gfx1250) — matrix operand order repaired and checked by reading, not compiled, not run on hardware
//
#include <hip/hip_runtime.h>
#include <stdint.h>

#define NN   4096   // N_NODES
#define NFEAT 48    // FEAT
#define GD   16     // GDIM

typedef __attribute__((ext_vector_type(16))) __bf16 v16bf;
typedef __attribute__((ext_vector_type(8)))  float  v8f;

struct alignas(32) BV { uint4 lo, hi; };

__device__ __forceinline__ unsigned short f2bf(float f) {
  unsigned int u = __float_as_uint(f);
  u += 0x7FFFu + ((u >> 16) & 1u);          // round-to-nearest-even
  return (unsigned short)(u >> 16);
}

__device__ __forceinline__ void atomicMaxF(float* a, float v) {
  if (v >= 0.f) atomicMax((int*)a, __float_as_int(v));
  else          atomicMin((unsigned int*)a, __float_as_uint(v));
}

// ---------------------------------------------------------------------------
// Generic linear: out = act(in @ W.T + b [+ resid]); act: 0=none, 1=sigmoid
// ---------------------------------------------------------------------------
__global__ void k_linear(const float* __restrict__ in, const float* __restrict__ W,
                         const float* __restrict__ b, const float* __restrict__ resid,
                         float* __restrict__ out, int rows, int in_f, int out_f, int act) {
  int i = blockIdx.x * blockDim.x + threadIdx.x;
  if (i >= rows * out_f) return;
  int r = i / out_f, o = i - r * out_f;
  const float* ip = in + (size_t)r * in_f;
  const float* wp = W + (size_t)o * in_f;
  float acc = b ? b[o] : 0.f;
  for (int k = 0; k < in_f; ++k) acc += ip[k] * wp[k];
  if (resid) acc += resid[i];
  if (act == 1) acc = 1.f / (1.f + __expf(-acc));
  out[i] = acc;
}

// ---------------------------------------------------------------------------
// lm = softmax(relu(src_emb @ tgt_emb), axis=1), stored bf16. One block per row.
// ---------------------------------------------------------------------------
__global__ void k_lm(const float* __restrict__ src_emb, const float* __restrict__ tgt_emb,
                     unsigned short* __restrict__ lm) {
  __shared__ float se[12];
  __shared__ float red[256];
  int row = blockIdx.x;
  if (threadIdx.x < 12) se[threadIdx.x] = src_emb[row * 12 + threadIdx.x];
  __syncthreads();
  float v[16];
  float mx = 0.f;                     // relu output >= 0
  for (int i = 0; i < 16; ++i) {
    int col = threadIdx.x * 16 + i;
    float d = 0.f;
    for (int k = 0; k < 12; ++k) d += se[k] * tgt_emb[k * NN + col];
    d = d > 0.f ? d : 0.f;
    v[i] = d;
    mx = fmaxf(mx, d);
  }
  red[threadIdx.x] = mx; __syncthreads();
  for (int s = 128; s > 0; s >>= 1) {
    if (threadIdx.x < s) red[threadIdx.x] = fmaxf(red[threadIdx.x], red[threadIdx.x + s]);
    __syncthreads();
  }
  mx = red[0]; __syncthreads();
  float sum = 0.f;
  for (int i = 0; i < 16; ++i) { v[i] = __expf(v[i] - mx); sum += v[i]; }
  red[threadIdx.x] = sum; __syncthreads();
  for (int s = 128; s > 0; s >>= 1) {
    if (threadIdx.x < s) red[threadIdx.x] += red[threadIdx.x + s];
    __syncthreads();
  }
  float inv = 1.f / red[0];
  unsigned short* op = lm + (size_t)row * NN + threadIdx.x * 16;
  for (int i = 0; i < 16; ++i) op[i] = f2bf(v[i] * inv);
}

// ---------------------------------------------------------------------------
// Transpose + bf16-convert node features: xt[b*f+j][k] = in[b*NN+k][j]
// ---------------------------------------------------------------------------
__global__ void k_xt(const float* __restrict__ in, unsigned short* __restrict__ xt,
                     int f, int Ntot) {
  int idx = blockIdx.x * blockDim.x + threadIdx.x;
  if (idx >= Ntot * NN) return;
  int col = idx / NN, k = idx - col * NN;
  int b = col / f, j = col - b * f;
  xt[(size_t)col * NN + k] = f2bf(in[((size_t)b * NN + k) * f + j]);
}

// ---------------------------------------------------------------------------
// y[b*NN+m][j] = sum_k lm[m][k] * X[b][k][j]   via v_wmma_f32_16x16x32_bf16.
// Each wave owns one 16-row M tile and a PAIR of N tiles (16x32 output):
// the A fragment (the only large operand, from lm) is loaded once and fed to
// two WMMAs per K-step, halving lm bandwidth. 4 M-tiles per 128-thread block.
// A lane layout (contiguous K): lane<16 -> K [k0..k0+15];
// lane>=16 -> [k0+16..k0+31]. B from Xt rows, same order.
// ---------------------------------------------------------------------------
__global__ void k_adp_wmma(const unsigned short* __restrict__ lm,
                           const unsigned short* __restrict__ xt,
                           float* __restrict__ y, int f) {
  int wave = threadIdx.x >> 5;
  int lane = threadIdx.x & 31;
  int mt = blockIdx.x * 4 + wave;
  int np = blockIdx.y;                     // N-tile pair index (covers cols np*32..+31)
  int grp = lane >> 4;
  int l15 = lane & 15;
  const unsigned short* arow  = lm + (size_t)(mt * 16 + l15) * NN;
  const unsigned short* brow0 = xt + (size_t)(np * 32 + l15) * NN;
  const unsigned short* brow1 = brow0 + (size_t)16 * NN;
  v8f acc0 = {}, acc1 = {};
  for (int k0 = 0; k0 < NN; k0 += 32) {
    const unsigned short* ap = arow + k0 + grp * 16;
    BV t;
    t.lo = *(const uint4*)ap;
    t.hi = *(const uint4*)(ap + 8);
    v16bf av  = __builtin_bit_cast(v16bf, t);
    v16bf bv0 = *(const v16bf*)(brow0 + k0 + grp * 16);
    v16bf bv1 = *(const v16bf*)(brow1 + k0 + grp * 16);
    acc0 = __builtin_amdgcn_wmma_f32_16x16x32_bf16(false, av, false, bv0,
                                                   (short)0, acc0, false, false);
    acc1 = __builtin_amdgcn_wmma_f32_16x16x32_bf16(false, av, false, bv1,
                                                   (short)0, acc1, false, false);
  }
  int mbase = mt * 16 + grp * 8;
  {
    int col = np * 32 + l15;
    int b = col / f, j = col - b * f;
    for (int r = 0; r < 8; ++r)
      y[(size_t)(b * NN + mbase + r) * f + j] = acc0[r];
  }
  {
    int col = np * 32 + 16 + l15;
    int b = col / f, j = col - b * f;
    for (int r = 0; r < 8; ++r)
      y[(size_t)(b * NN + mbase + r) * f + j] = acc1[r];
  }
}

// ---------------------------------------------------------------------------
// GAT support kernels
// ---------------------------------------------------------------------------
__global__ void k_attn(const float* __restrict__ h, const float* __restrict__ a_src,
                       const float* __restrict__ a_dst, float* __restrict__ as,
                       float* __restrict__ ad, int n, int H) {
  int i = blockIdx.x * blockDim.x + threadIdx.x;
  if (i >= n * H) return;
  int node = i / H, hh = i - node * H;
  const float* hp = h + ((size_t)node * H + hh) * GD;
  float s = 0.f, d = 0.f;
  for (int c = 0; c < GD; ++c) {
    s += hp[c] * a_src[hh * GD + c];
    d += hp[c] * a_dst[hh * GD + c];
  }
  as[i] = s; ad[i] = d;
}

__global__ void k_gat_init(float* __restrict__ m, float* __restrict__ z,
                           float* __restrict__ acc, int n, int H) {
  int i = blockIdx.x * blockDim.x + threadIdx.x;
  if (i >= n * H * GD) return;
  acc[i] = 0.f;
  if (i < n * H) { m[i] = -1e30f; z[i] = 0.f; }
}

__device__ __forceinline__ void edge_sd(const int* ei, int E, int e, int& s, int& d) {
  if (e < E) { s = ei[e]; d = ei[E + e]; }
  else       { s = e - E; d = e - E; }      // self-loops appended
}

__global__ void k_edge_max(const int* __restrict__ ei, int E, int n,
                           const float* __restrict__ as, const float* __restrict__ ad,
                           float* __restrict__ m, int H) {
  int e = blockIdx.x * blockDim.x + threadIdx.x;
  if (e >= E + n) return;
  int s, d; edge_sd(ei, E, e, s, d);
  for (int h = 0; h < H; ++h) {
    float v = as[s * H + h] + ad[d * H + h];
    v = v > 0.f ? v : 0.2f * v;               // leaky_relu(., 0.2)
    atomicMaxF(&m[d * H + h], v);
  }
}

__global__ void k_edge_sum(const int* __restrict__ ei, int E, int n,
                           const float* __restrict__ as, const float* __restrict__ ad,
                           const float* __restrict__ m, float* __restrict__ z, int H) {
  int e = blockIdx.x * blockDim.x + threadIdx.x;
  if (e >= E + n) return;
  int s, d; edge_sd(ei, E, e, s, d);
  for (int h = 0; h < H; ++h) {
    float v = as[s * H + h] + ad[d * H + h];
    v = v > 0.f ? v : 0.2f * v;
    atomicAdd(&z[d * H + h], __expf(v - m[d * H + h]));
  }
}

__global__ void k_edge_acc(const int* __restrict__ ei, int E, int n,
                           const float* __restrict__ as, const float* __restrict__ ad,
                           const float* __restrict__ m, const float* __restrict__ z,
                           const float* __restrict__ hfeat, float* __restrict__ acc, int H) {
  int e = blockIdx.x * blockDim.x + threadIdx.x;
  if (e >= E + n) return;
  int s, d; edge_sd(ei, E, e, s, d);
  for (int h = 0; h < H; ++h) {
    float v = as[s * H + h] + ad[d * H + h];
    v = v > 0.f ? v : 0.2f * v;
    float alpha = __expf(v - m[d * H + h]) / z[d * H + h];
    const float* hp = hfeat + ((size_t)s * H + h) * GD;
    float* op = acc + ((size_t)d * H + h) * GD;
    for (int c = 0; c < GD; ++c) atomicAdd(&op[c], alpha * hp[c]);
  }
}

__global__ void k_gat_final(const float* __restrict__ acc, const float* __restrict__ b,
                            float* __restrict__ g, int n, int H) {
  int i = blockIdx.x * blockDim.x + threadIdx.x;
  if (i >= n * GD) return;
  int node = i / GD, c = i - node * GD;
  float s = 0.f;
  for (int h = 0; h < H; ++h) s += acc[((size_t)node * H + h) * GD + c];
  g[i] = s / (float)H + b[c];
}

// out = act(g)*s + alt*(1-s); mode: 0=tanh, 1=leaky(0.01), 2=relu
__global__ void k_combine(const float* __restrict__ g, const float* __restrict__ s,
                          const float* __restrict__ alt, float* __restrict__ out,
                          int cnt, int mode) {
  int i = blockIdx.x * blockDim.x + threadIdx.x;
  if (i >= cnt) return;
  float gv = g[i];
  if (mode == 0)      gv = tanhf(gv);
  else if (mode == 1) gv = gv > 0.f ? gv : 0.01f * gv;
  else                gv = gv > 0.f ? gv : 0.f;
  float sv = s[i];
  out[i] = gv * sv + alt[i] * (1.f - sv);
}

// mode: 0=tanh, 1=relu
__global__ void k_act(const float* __restrict__ in, float* __restrict__ out, int cnt, int mode) {
  int i = blockIdx.x * blockDim.x + threadIdx.x;
  if (i >= cnt) return;
  float v = in[i];
  out[i] = mode == 0 ? tanhf(v) : (v > 0.f ? v : 0.f);
}

__global__ void k_concat(const float* __restrict__ a, const float* __restrict__ b,
                         float* __restrict__ out, int rows) {
  int i = blockIdx.x * blockDim.x + threadIdx.x;
  if (i >= rows * 2 * GD) return;
  int r = i / (2 * GD), c = i - r * 2 * GD;
  out[i] = c < GD ? a[r * GD + c] : b[r * GD + (c - GD)];
}

// ---------------------------------------------------------------------------
// Host orchestration
// ---------------------------------------------------------------------------
static inline int cdiv(long a, long b) { return (int)((a + b - 1) / b); }

extern "C" void kernel_launch(void* const* d_in, const int* in_sizes, int n_in,
                              void* d_out, int out_size, void* d_ws, size_t ws_size,
                              hipStream_t stream) {
  const float* x  = (const float*)d_in[0];
  const int*  ei  = (const int*)d_in[1];
  const int*  dei = (const int*)d_in[2];
  const int n = in_sizes[0] / NFEAT;      // 8192
  const int E = in_sizes[1] / 2;          // 262144
  const int B = n / NN;                   // 2

  auto P = [&](int i) { return (const float*)d_in[i]; };

  // workspace carve-up
  char* wsp = (char*)d_ws;
  size_t off = 0;
  auto wsa = [&](size_t bytes) -> char* {
    char* p = wsp + off;
    off = (off + bytes + 255) & ~(size_t)255;
    return p;
  };
  unsigned short* lmb = (unsigned short*)wsa((size_t)NN * NN * 2);
  unsigned short* xtb = (unsigned short*)wsa((size_t)B * NFEAT * NN * 2);
  float* yb   = (float*)wsa((size_t)n * NFEAT * 4);
  float* x1   = (float*)wsa((size_t)n * NFEAT * 4);
  float* x2   = (float*)wsa((size_t)n * NFEAT * 4);
  float* hb   = (float*)wsa((size_t)n * 3 * GD * 4);
  float* asb  = (float*)wsa((size_t)n * 3 * 4);
  float* adb  = (float*)wsa((size_t)n * 3 * 4);
  float* mb   = (float*)wsa((size_t)n * 3 * 4);
  float* zb   = (float*)wsa((size_t)n * 3 * 4);
  float* accb = (float*)wsa((size_t)n * 3 * GD * 4);
  float* gb   = (float*)wsa((size_t)n * GD * 4);
  float* sb   = (float*)wsa((size_t)n * GD * 4);
  float* oA   = (float*)wsa((size_t)n * GD * 4);
  float* oB   = (float*)wsa((size_t)n * GD * 4);
  float* tb   = (float*)wsa((size_t)n * GD * 4);
  float* orig = (float*)wsa((size_t)n * GD * 4);
  float* spo  = (float*)wsa((size_t)n * GD * 4);
  float* dtwo = (float*)wsa((size_t)n * GD * 4);
  float* cell = (float*)wsa((size_t)n * 2 * GD * 4);

  auto linear = [&](const float* in, const float* W, const float* b, const float* resid,
                    float* out, int rows, int inf, int outf, int act) {
    int tot = rows * outf;
    k_linear<<<cdiv(tot, 256), 256, 0, stream>>>(in, W, b, resid, out, rows, inf, outf, act);
  };

  auto gat = [&](const float* xin, int inf, int H, int pb, const int* edges, float* g) {
    linear(xin, P(pb + 0), nullptr, nullptr, hb, n, inf, H * GD, 0);
    k_attn<<<cdiv((long)n * H, 256), 256, 0, stream>>>(hb, P(pb + 1), P(pb + 2), asb, adb, n, H);
    k_gat_init<<<cdiv((long)n * H * GD, 256), 256, 0, stream>>>(mb, zb, accb, n, H);
    int ET = E + n;
    k_edge_max<<<cdiv(ET, 256), 256, 0, stream>>>(edges, E, n, asb, adb, mb, H);
    k_edge_sum<<<cdiv(ET, 256), 256, 0, stream>>>(edges, E, n, asb, adb, mb, zb, H);
    k_edge_acc<<<cdiv(ET, 256), 256, 0, stream>>>(edges, E, n, asb, adb, mb, zb, hb, accb, H);
    k_gat_final<<<cdiv((long)n * GD, 256), 256, 0, stream>>>(accb, P(pb + 3), g, n, H);
  };

  auto adp = [&](const float* xin, int f, const float* lw, const float* lb, float* sout) {
    int Ntot = B * f;                                   // 96 or 32 (32-divisible)
    k_xt<<<cdiv((long)Ntot * NN, 256), 256, 0, stream>>>(xin, xtb, f, Ntot);
    dim3 grid(NN / 16 / 4, Ntot / 32);
    k_adp_wmma<<<grid, 128, 0, stream>>>(lmb, xtb, yb, f);
    linear(yb, lw, lb, nullptr, sout, n, f, GD, 1);     // sigmoid
  };

  auto branch = [&](const float* xin, const int* edges, int pb, int h4, float* bout) {
    k_lm<<<NN, 256, 0, stream>>>(P(pb + 0), P(pb + 1), lmb);
    int cnt = n * GD;
    // layer 1
    gat(xin, NFEAT, 3, pb + 2, edges, gb);
    adp(xin, NFEAT, P(pb + 20), P(pb + 21), sb);
    linear(xin, P(pb + 18), P(pb + 19), nullptr, orig, n, NFEAT, GD, 0);
    k_combine<<<cdiv(cnt, 256), 256, 0, stream>>>(gb, sb, orig, oA, cnt, 0);
    // layer 2
    k_act<<<cdiv(cnt, 256), 256, 0, stream>>>(oA, tb, cnt, 0);
    gat(tb, GD, 3, pb + 6, edges, gb);
    adp(tb, GD, P(pb + 22), P(pb + 23), sb);
    k_combine<<<cdiv(cnt, 256), 256, 0, stream>>>(gb, sb, oA, oB, cnt, 1);
    // layer 3
    k_act<<<cdiv(cnt, 256), 256, 0, stream>>>(oB, tb, cnt, 1);
    gat(tb, GD, 3, pb + 10, edges, gb);
    adp(tb, GD, P(pb + 24), P(pb + 25), sb);
    k_combine<<<cdiv(cnt, 256), 256, 0, stream>>>(gb, sb, oB, oA, cnt, 2);
    // layer 4
    k_act<<<cdiv(cnt, 256), 256, 0, stream>>>(oA, tb, cnt, 1);
    gat(tb, GD, h4, pb + 14, edges, gb);
    adp(tb, GD, P(pb + 26), P(pb + 27), sb);
    k_combine<<<cdiv(cnt, 256), 256, 0, stream>>>(gb, sb, oA, bout, cnt, 2);
  };

  // x1 = x @ seqW.T + seqb + x ; branches ; concat ; out
  linear(x, P(3), P(4), x, x1, n, NFEAT, NFEAT, 0);
  branch(x1, ei, 7, 1, spo);
  linear(x1, P(3), P(4), x1, x2, n, NFEAT, NFEAT, 0);
  branch(x2, dei, 35, 3, dtwo);
  k_concat<<<cdiv((long)n * 2 * GD, 256), 256, 0, stream>>>(spo, dtwo, cell, n);
  linear(cell, P(5), P(6), nullptr, (float*)d_out, n, 2 * GD, NFEAT, 0);
}
